// SSTv2_35124242546748
// MI455X (gfx1250) — matrix operand order repaired and checked by reading, not compiled, not run on hardware
//
#include <hip/hip_runtime.h>
#include <hip/hip_bf16.h>
#include <math.h>

// ---- problem constants (match reference) ----
#define NPTS   30000
#define CDIM   128
#define NHEADS 8
#define DHEAD  16
#define FFDIM  256
#define NLAYERS 12
#define NWIN   400
#define WSLOT  100
#define FILLN  75
#define PWS    80                 // padded (compacted) slots per window, 5x16
#define SLOTS  (NWIN * PWS)       // 32000
#define BB 2
#define HH 400
#define WW 400

typedef __attribute__((ext_vector_type(16))) __bf16 v16bf;
typedef __attribute__((ext_vector_type(8)))  __bf16 v8bf;
typedef __attribute__((ext_vector_type(4)))  __bf16 v4bf;
typedef __attribute__((ext_vector_type(8)))  float  v8f;

// float -> bf16: backend uses native v_cvt_pk_bf16_f32
__device__ __forceinline__ __bf16 f2bf(float f) { return (__bf16)f; }

__device__ __forceinline__ v8bf cvt8(float4 a, float4 b) {
  v8bf r;
  r[0] = f2bf(a.x); r[1] = f2bf(a.y); r[2] = f2bf(a.z); r[3] = f2bf(a.w);
  r[4] = f2bf(b.x); r[5] = f2bf(b.y); r[6] = f2bf(b.z); r[7] = f2bf(b.w);
  return r;
}

__device__ __forceinline__ v16bf combine(v8bf lo, v8bf hi) {
  return __builtin_shufflevector(lo, hi, 0, 1, 2, 3, 4, 5, 6, 7, 8, 9, 10, 11,
                                 12, 13, 14, 15);
}

__device__ __forceinline__ v8f wmma_bf16(v16bf a, v16bf b, v8f c) {
  return __builtin_amdgcn_wmma_f32_16x16x32_bf16(false, a, false, b, (short)0, c,
                                                 false, false);
}

// async 16B global->LDS copy (per-lane), tracked by ASYNCcnt
__device__ __forceinline__ void async_b128(void* lds, const void* gaddr) {
  unsigned dst = (unsigned)(size_t)lds;  // LDS aperture: low 32 bits = offset
  asm volatile("global_load_async_to_lds_b128 %0, %1, off"
               :
               : "v"(dst), "v"(gaddr)
               : "memory");
}
__device__ __forceinline__ void wait_async0() {
  asm volatile("s_wait_asynccnt 0" ::: "memory");
}

// CDNA5 16-bit fragment layouts (wave32):
//  A (16x32 MxK): lane m = lane&15, kb8 = (lane&16)?8:0;
//      e<8 -> k = kb8+e ; e>=8 -> k = 16+kb8+(e-8)   (two runs of 8)
//  B (32x16 KxN): lane n = lane&15, k per e same as A (kb8 runs of 8)
//  C/D f32 16x16: element r -> m = r + ((lane&16)?8:0), n = lane&15

#define A_STRIDE 36  // f32 row stride (144B, 16B-aligned, bank-spread)
#define B_STRIDE 40  // bf16 row stride (80B, 16B-aligned, bank-spread)

// ------------------------------------------------------------------
// GEMM: C[M,N] = act(A[M,K] @ Bw^T + bias); Bw is bf16 [N][K] (pre-packed).
// Block: 256 threads (8 waves), tile 128x64, K-step 32, double-buffered
// async global->LDS staging (A as f32, B as bf16).
// ------------------------------------------------------------------
template <int ACT>
__global__ __launch_bounds__(256) void gemm_kernel(
    const float* __restrict__ A, const __bf16* __restrict__ Bw,
    const float* __restrict__ bias, float* __restrict__ C,
    int M, int N, int K) {
  __shared__ float Af[2][128 * A_STRIDE];   // 36864 B
  __shared__ __bf16 Bb[2][64 * B_STRIDE];   // 10240 B
  const int tid = threadIdx.x;
  const int lane = tid & 31;
  const int wave = tid >> 5;
  const int m0 = blockIdx.y * 128;
  const int n0 = blockIdx.x * 64;
  const int kb8 = (lane & 16) ? 8 : 0;
  const int kb16 = (lane & 16) ? 16 : 0;

  v8f acc[4] = {};

  auto prefetch = [&](int buf, int k0) {
    // A tile: 128 rows x 128B = 1024 16B chunks, 4 per thread
#pragma unroll
    for (int j = 0; j < 4; ++j) {
      int c = tid + j * 256;
      int row = c >> 3, ch = c & 7;
      int gm = m0 + row;
      if (gm < M)
        async_b128(&Af[buf][row * A_STRIDE + ch * 4],
                   &A[(size_t)gm * K + k0 + ch * 4]);
    }
    // B tile: 64 rows x 64B = 256 16B chunks, 1 per thread
    {
      int row = tid >> 2, ch = tid & 3;
      int gn = n0 + row;
      if (gn < N)
        async_b128(&Bb[buf][row * B_STRIDE + ch * 8],
                   &Bw[(size_t)gn * K + k0 + ch * 8]);
    }
  };

  const int nk = K >> 5;
  prefetch(0, 0);
  for (int ki = 0; ki < nk; ++ki) {
    const int buf = ki & 1;
    wait_async0();
    __syncthreads();
    if (ki + 1 < nk) prefetch(buf ^ 1, (ki + 1) << 5);

    const int arow = wave * 16 + (lane & 15);
    const float4* ap = (const float4*)&Af[buf][arow * A_STRIDE + kb8];
    const float4* ap2 = (const float4*)&Af[buf][arow * A_STRIDE + 16 + kb8];
    v16bf af = combine(cvt8(ap[0], ap[1]), cvt8(ap2[0], ap2[1]));
#pragma unroll
    for (int t = 0; t < 4; ++t) {
      const int bn = t * 16 + (lane & 15);
      v16bf bf_ = combine(*(const v8bf*)&Bb[buf][bn * B_STRIDE + kb8],
                          *(const v8bf*)&Bb[buf][bn * B_STRIDE + kb8 + 16]);
      acc[t] = wmma_bf16(af, bf_, acc[t]);
    }
  }

#pragma unroll
  for (int t = 0; t < 4; ++t) {
    int n = n0 + t * 16 + (lane & 15);
    float bv = (bias && n < N) ? bias[n] : 0.f;
#pragma unroll
    for (int r = 0; r < 8; ++r) {
      int m = m0 + wave * 16 + r + ((lane & 16) ? 8 : 0);
      if (m < M && n < N) {
        float v = acc[t][r] + bv;
        if (ACT == 1) v = 0.5f * v * (1.0f + erff(v * 0.70710678118654752f));
        C[(size_t)m * N + n] = v;
      }
    }
  }
}

// ------------------------------------------------------------------
// Windowed attention: block = one window (256 thr, 8 waves = 8 heads).
// ------------------------------------------------------------------
__global__ __launch_bounds__(256) void attn_kernel(
    const float* __restrict__ QK, const float* __restrict__ V,
    float* __restrict__ O) {
  __shared__ float P[NHEADS][16][84];     // 43008 B
  __shared__ __bf16 VT[NHEADS][16][80];   // 20480 B, [head][n][k]
  const int lane = threadIdx.x & 31;
  const int h = threadIdx.x >> 5;
  const int kb8 = (lane & 16) ? 8 : 0;
  const int kb16 = (lane & 16) ? 16 : 0;
  const size_t base = (size_t)blockIdx.x * PWS;

  for (int i = threadIdx.x; i < NHEADS * 80 * 16; i += 256) {
    int n = i & 15;
    int k = (i >> 4) % 80;
    int hh = i / (80 * 16);
    VT[hh][n][k] = f2bf(V[(base + k) * 128 + hh * DHEAD + n]);
  }
  __syncthreads();

  const v8bf z8 = {};
  for (int rb = 0; rb < 5; ++rb) {
    // Q fragment (1/sqrt(DH)=0.25 folded in); e>=8 statically zero
    const float4* qp =
        (const float4*)&QK[(base + rb * 16 + (lane & 15)) * 256 + h * DHEAD + kb8];
    float4 q0 = qp[0], q1 = qp[1];
    q0.x *= 0.25f; q0.y *= 0.25f; q0.z *= 0.25f; q0.w *= 0.25f;
    q1.x *= 0.25f; q1.y *= 0.25f; q1.z *= 0.25f; q1.w *= 0.25f;
    v16bf qf = combine(cvt8(q0, q1), z8);

#pragma unroll
    for (int ct = 0; ct < 5; ++ct) {
      v16bf kf = {};
      {
        const float4* kp =
            (const float4*)&QK[(base + ct * 16 + (lane & 15)) * 256 + 128 + h * DHEAD + kb8];
        kf = combine(cvt8(kp[0], kp[1]), z8);
      }
      v8f z = {};
      v8f s = wmma_bf16(qf, kf, z);
#pragma unroll
      for (int r = 0; r < 8; ++r) {
        int m = r + ((lane & 16) ? 8 : 0);
        P[h][m][ct * 16 + (lane & 15)] = s[r];
      }
    }
    __syncthreads();
    if (lane < 16) {
      float mx = -1e30f;
      for (int c = 0; c < FILLN; ++c) mx = fmaxf(mx, P[h][lane][c]);
      float sum = 0.f;
      for (int c = 0; c < FILLN; ++c) {
        float e = expf(P[h][lane][c] - mx);
        P[h][lane][c] = e;
        sum += e;
      }
      float inv = 1.0f / sum;
      for (int c = 0; c < FILLN; ++c) P[h][lane][c] *= inv;
      for (int c = FILLN; c < 80; ++c) P[h][lane][c] = 0.f;
    }
    __syncthreads();

    v8f o = {};
#pragma unroll
    for (int kb = 0; kb < 96; kb += 32) {
      const float4* plo = (const float4*)&P[h][lane & 15][kb + kb8];
      v8bf lo = cvt8(plo[0], plo[1]);
      v8bf hi = z8;
      if (kb < 64) {
        const float4* phi = (const float4*)&P[h][lane & 15][kb + 16 + kb8];
        hi = cvt8(phi[0], phi[1]);
      }
      v16bf pf = combine(lo, hi);

      v16bf vf = {};
      {
        v8bf vlo = *(const v8bf*)&VT[h][lane & 15][kb + kb8];
        v8bf vhi = z8;
        if (kb < 64) vhi = *(const v8bf*)&VT[h][lane & 15][kb + 16 + kb8];
        vf = combine(vlo, vhi);
      }
      o = wmma_bf16(pf, vf, o);
    }
#pragma unroll
    for (int r = 0; r < 8; ++r) {
      int m = r + ((lane & 16) ? 8 : 0);
      O[(base + rb * 16 + m) * 128 + h * DHEAD + (lane & 15)] = o[r];
    }
    __syncthreads();
  }
}

// ------------------------------------------------------------------
__global__ void scatter_kernel(const float* __restrict__ x,
                               const float* __restrict__ pos,
                               const int* __restrict__ inds,
                               float* __restrict__ qkb, float* __restrict__ vb) {
  int i = blockIdx.x, lane = threadIdx.x;  // 32 threads
  int slot = inds[i];
  size_t cs = (size_t)(slot / WSLOT) * PWS + (slot % WSLOT);
  float4 xv = ((const float4*)(x + (size_t)i * 128))[lane];
  float4 pv = ((const float4*)(pos + (size_t)i * 128))[lane];
  float4 q = {xv.x + pv.x, xv.y + pv.y, xv.z + pv.z, xv.w + pv.w};
  ((float4*)(qkb + cs * 128))[lane] = q;
  ((float4*)(vb + cs * 128))[lane] = xv;
}

__global__ void zero_pad_kernel(float* __restrict__ qkb, float* __restrict__ vb) {
  int idx = blockIdx.x * blockDim.x + threadIdx.x;  // NWIN*5*32 float4s
  if (idx >= NWIN * 5 * 32) return;
  int q = idx & 31;
  int t = idx >> 5;
  size_t row = (size_t)(t / 5) * PWS + FILLN + (t % 5);
  float4 z = {0.f, 0.f, 0.f, 0.f};
  ((float4*)(qkb + row * 128))[q] = z;
  ((float4*)(vb + row * 128))[q] = z;
}

// ------------------------------------------------------------------
template <bool GATHER>
__global__ __launch_bounds__(256) void res_ln_kernel(
    const float* __restrict__ xin, const float* __restrict__ add,
    const int* __restrict__ inds, const float* __restrict__ g,
    const float* __restrict__ b, float* __restrict__ xout) {
  int pt = blockIdx.x * 8 + (threadIdx.x >> 5);
  if (pt >= NPTS) return;
  int lane = threadIdx.x & 31;
  size_t arow;
  if (GATHER) {
    int slot = inds[pt];
    arow = (size_t)(slot / WSLOT) * PWS + (slot % WSLOT);
  } else {
    arow = (size_t)pt;
  }
  float4 xv = ((const float4*)(xin + (size_t)pt * 128))[lane];
  float4 av = ((const float4*)(add + arow * 128))[lane];
  float va[4] = {xv.x + av.x, xv.y + av.y, xv.z + av.z, xv.w + av.w};
  float s = va[0] + va[1] + va[2] + va[3];
  float s2 = va[0] * va[0] + va[1] * va[1] + va[2] * va[2] + va[3] * va[3];
#pragma unroll
  for (int o = 16; o > 0; o >>= 1) {
    s += __shfl_xor(s, o, 32);
    s2 += __shfl_xor(s2, o, 32);
  }
  float mean = s * (1.f / 128.f);
  float var = s2 * (1.f / 128.f) - mean * mean;
  float rstd = rsqrtf(var + 1e-5f);
  float4 gv = ((const float4*)g)[lane];
  float4 bv = ((const float4*)b)[lane];
  float4 ov = {(va[0] - mean) * rstd * gv.x + bv.x,
               (va[1] - mean) * rstd * gv.y + bv.y,
               (va[2] - mean) * rstd * gv.z + bv.z,
               (va[3] - mean) * rstd * gv.w + bv.w};
  ((float4*)(xout + (size_t)pt * 128))[lane] = ov;
}

__global__ void fill_kernel(float* __restrict__ p, long long n) {
  long long i = (long long)blockIdx.x * blockDim.x + threadIdx.x;
  long long stride = (long long)gridDim.x * blockDim.x;
  for (; i < n; i += stride) p[i] = 0.f;
}

// canvas is NHWC: fully coalesced float4 scatter
__global__ void bev_scatter_kernel(const float* __restrict__ x,
                                   const int* __restrict__ coors,
                                   float* __restrict__ canvas) {
  int i = blockIdx.x, lane = threadIdx.x;  // 32 threads
  int b = coors[i * 4 + 0];
  int y = coors[i * 4 + 2];
  int xx = coors[i * 4 + 3];
  float4 v = ((const float4*)(x + (size_t)i * 128))[lane];
  ((float4*)(canvas + (((size_t)b * HH + y) * WW + xx) * 128))[lane] = v;
}

// ---- one-time weight packing: f32 -> bf16 [N][K] -----------------
// direct copy (src already [.., N, K] row-major)
__global__ void prep_bf16_kernel(const float* __restrict__ src,
                                 __bf16* __restrict__ dst, int n) {
  int i = blockIdx.x * 256 + threadIdx.x;
  if (i < n) dst[i] = f2bf(src[i]);
}
// per-layer transpose: src [L][Kd][Nd] -> dst [L][Nd][Kd]
__global__ void prep_tr_kernel(const float* __restrict__ src,
                               __bf16* __restrict__ dst, int Kd, int Nd) {
  int i = blockIdx.x * 256 + threadIdx.x;
  if (i >= NLAYERS * Kd * Nd) return;
  int k = i % Kd;
  int n = (i / Kd) % Nd;
  int l = i / (Kd * Nd);
  dst[i] = f2bf(src[((size_t)l * Kd + k) * Nd + n]);
}

// conv weights f32 OIHW -> bf16 [conv][tap][oc][ic]
__global__ void prep_w_kernel(const float* __restrict__ W,
                              __bf16* __restrict__ Wb) {
  int idx = blockIdx.x * 256 + threadIdx.x;
  if (idx >= 2 * 9 * 128 * 128) return;
  int ic = idx & 127;
  int rest = idx >> 7;
  int oc = rest & 127; rest >>= 7;
  int kx = rest % 3; rest /= 3;
  int ky = rest % 3;
  int ci = rest / 3;
  Wb[idx] = f2bf(W[((((size_t)ci * 128 + oc) * 128 + ic) * 3 + ky) * 3 + kx]);
}

// ------------------------------------------------------------------
// Implicit-GEMM 3x3 dilated(2) conv, pad 2, fused BN+ReLU.
// ------------------------------------------------------------------
template <int NHWC_IN>
__global__ __launch_bounds__(256) void conv_kernel(
    const float* __restrict__ in, const __bf16* __restrict__ Wb,
    const float* __restrict__ bng, const float* __restrict__ bnb,
    const float* __restrict__ bnm, const float* __restrict__ bnv,
    float* __restrict__ out) {
  __shared__ __bf16 InT[3][20][136];  // 16320 B, [ky][col][ic]
  const int lane = threadIdx.x & 31;
  const int wave = threadIdx.x >> 5;
  const int kb8 = (lane & 16) ? 8 : 0;
  const int kb16 = (lane & 16) ? 16 : 0;
  int tile = blockIdx.x;
  const int xt = tile % (WW / 16);
  tile /= (WW / 16);
  const int y = tile % HH;
  const int b = tile / HH;
  const int oc0 = wave * 16;

  if (NHWC_IN) {
    for (int i = threadIdx.x; i < 3 * 20 * 32; i += 256) {
      int ic4 = (i & 31) * 4;
      int col = (i >> 5) % 20;
      int ky = i / (20 * 32);
      int iy = y + 2 * (ky - 1);
      int ix = xt * 16 - 2 + col;
      float4 v = {0.f, 0.f, 0.f, 0.f};
      if (iy >= 0 && iy < HH && ix >= 0 && ix < WW)
        v = *(const float4*)&in[(((size_t)b * HH + iy) * WW + ix) * 128 + ic4];
      v4bf bv;
      bv[0] = f2bf(v.x); bv[1] = f2bf(v.y); bv[2] = f2bf(v.z); bv[3] = f2bf(v.w);
      *(v4bf*)&InT[ky][col][ic4] = bv;
    }
  } else {
    for (int i = threadIdx.x; i < 3 * 20 * 128; i += 256) {
      int col = i % 20;
      int ic = (i / 20) & 127;
      int ky = i / (20 * 128);
      int iy = y + 2 * (ky - 1);
      int ix = xt * 16 - 2 + col;
      float v = (iy >= 0 && iy < HH && ix >= 0 && ix < WW)
                    ? in[(((size_t)b * 128 + ic) * HH + iy) * WW + ix]
                    : 0.f;
      InT[ky][col][ic] = f2bf(v);
    }
  }
  __syncthreads();

  v8f acc = {};
#pragma unroll
  for (int ky = 0; ky < 3; ++ky) {
#pragma unroll
    for (int kx = 0; kx < 3; ++kx) {
      const int col = (lane & 15) + 2 * kx;
#pragma unroll
      for (int icb = 0; icb < 4; ++icb) {
        size_t wbase = (((size_t)(ky * 3 + kx) * 128) + oc0 + (lane & 15)) * 128 +
                       icb * 32 + kb8;
        v16bf af = combine(*(const v8bf*)&Wb[wbase],
                           *(const v8bf*)&Wb[wbase + 16]);
        const v8bf* ip = (const v8bf*)&InT[ky][col][icb * 32 + kb8];
        v16bf bf_ = combine(ip[0], ip[2]);
        acc = wmma_bf16(af, bf_, acc);
      }
    }
  }
#pragma unroll
  for (int r = 0; r < 8; ++r) {
    int oc = oc0 + r + ((lane & 16) ? 8 : 0);
    int px = xt * 16 + (lane & 15);
    float sc = bng[oc] * rsqrtf(bnv[oc] + 1e-3f);
    float sh = bnb[oc] - bnm[oc] * sc;
    float v = fmaxf(acc[r] * sc + sh, 0.f);
    out[(((size_t)b * 128 + oc) * HH + y) * WW + px] = v;
  }
}

// ------------------------------------------------------------------
extern "C" void kernel_launch(void* const* d_in, const int* in_sizes, int n_in,
                              void* d_out, int out_size, void* d_ws,
                              size_t ws_size, hipStream_t stream) {
  (void)in_sizes; (void)n_in; (void)out_size; (void)ws_size;
  const float* voxel = (const float*)d_in[0];
  const float* pos0 = (const float*)d_in[1];
  const float* pos1 = (const float*)d_in[2];
  const int* coors = (const int*)d_in[3];
  const int* inds0 = (const int*)d_in[4];
  const int* inds1 = (const int*)d_in[5];
  const float* Wqkv = (const float*)d_in[8];
  const float* bqkv = (const float*)d_in[9];
  const float* Wo = (const float*)d_in[10];
  const float* bo = (const float*)d_in[11];
  const float* W1 = (const float*)d_in[12];
  const float* b1 = (const float*)d_in[13];
  const float* W2 = (const float*)d_in[14];
  const float* b2 = (const float*)d_in[15];
  const float* g1 = (const float*)d_in[16];
  const float* be1 = (const float*)d_in[17];
  const float* g2 = (const float*)d_in[18];
  const float* be2 = (const float*)d_in[19];
  const float* convW = (const float*)d_in[20];
  const float* bng = (const float*)d_in[21];
  const float* bnb = (const float*)d_in[22];
  const float* bnm = (const float*)d_in[23];
  const float* bnv = (const float*)d_in[24];

  float* Wf = (float*)d_ws;
  size_t off = 0;
  auto alloc = [&](size_t n) {
    float* p = Wf + off;
    off += (n + 3) & ~(size_t)3;  // keep 16B alignment
    return p;
  };
  float* xbuf = alloc((size_t)NPTS * 128);
  float* tbuf = alloc((size_t)NPTS * 128);
  float* qkb = alloc((size_t)SLOTS * 128);
  float* vb = alloc((size_t)SLOTS * 128);
  float* qkout = alloc((size_t)SLOTS * 256);
  float* vout = alloc((size_t)SLOTS * 128);
  float* obuf = alloc((size_t)SLOTS * 128);
  float* oproj = alloc((size_t)SLOTS * 128);
  float* ffh = alloc((size_t)NPTS * 256);
  float* ffo = alloc((size_t)NPTS * 128);
  float* canvas = alloc((size_t)BB * 128 * HH * WW);  // NHWC
  float* bev1 = alloc((size_t)BB * 128 * HH * WW);    // NCHW
  __bf16* Wbconv = (__bf16*)alloc((size_t)2 * 9 * 128 * 128 / 2);
  __bf16* WqkvB = (__bf16*)alloc((size_t)NLAYERS * 384 * 128 / 2);  // [N][K]
  __bf16* WoB = (__bf16*)alloc((size_t)NLAYERS * 128 * 128 / 2);    // [N][K]
  __bf16* W1T = (__bf16*)alloc((size_t)NLAYERS * 256 * 128 / 2);    // [256][128]
  __bf16* W2T = (__bf16*)alloc((size_t)NLAYERS * 128 * 256 / 2);    // [128][256]

  // one-time weight packing (deterministic, replayed every launch)
  {
    int nq = NLAYERS * 384 * 128;
    prep_bf16_kernel<<<(nq + 255) / 256, 256, 0, stream>>>(Wqkv, WqkvB, nq);
    int no = NLAYERS * 128 * 128;
    prep_bf16_kernel<<<(no + 255) / 256, 256, 0, stream>>>(Wo, WoB, no);
    int n1 = NLAYERS * 128 * 256;
    prep_tr_kernel<<<(n1 + 255) / 256, 256, 0, stream>>>(W1, W1T, 128, 256);
    prep_tr_kernel<<<(n1 + 255) / 256, 256, 0, stream>>>(W2, W2T, 256, 128);
    prep_w_kernel<<<(2 * 9 * 128 * 128 + 255) / 256, 256, 0, stream>>>(convW,
                                                                       Wbconv);
  }

  const float* cur = voxel;
  for (int l = 0; l < NLAYERS; ++l) {
    const float* pos = (l & 1) ? pos1 : pos0;
    const int* inds = (l & 1) ? inds1 : inds0;

    scatter_kernel<<<NPTS, 32, 0, stream>>>(cur, pos, inds, qkb, vb);
    zero_pad_kernel<<<(NWIN * 5 * 32 + 255) / 256, 256, 0, stream>>>(qkb, vb);

    // Q|K = qk @ Wqkv[:256].T + bqkv[:256]
    gemm_kernel<0><<<dim3(4, SLOTS / 128), 256, 0, stream>>>(
        qkb, WqkvB + (size_t)l * 384 * 128, bqkv + (size_t)l * 384, qkout,
        SLOTS, 256, 128);
    // V = win @ Wqkv[256:].T + bqkv[256:]
    gemm_kernel<0><<<dim3(2, SLOTS / 128), 256, 0, stream>>>(
        vb, WqkvB + (size_t)l * 384 * 128 + (size_t)256 * 128,
        bqkv + (size_t)l * 384 + 256, vout, SLOTS, 128, 128);

    attn_kernel<<<NWIN, 256, 0, stream>>>(qkout, vout, obuf);

    // out projection
    gemm_kernel<0><<<dim3(2, SLOTS / 128), 256, 0, stream>>>(
        obuf, WoB + (size_t)l * 128 * 128, bo + (size_t)l * 128, oproj,
        SLOTS, 128, 128);

    // gather + residual + LN1
    res_ln_kernel<true><<<(NPTS + 7) / 8, 256, 0, stream>>>(
        cur, oproj, inds, g1 + l * 128, be1 + l * 128, tbuf);

    // FFN
    gemm_kernel<1><<<dim3(4, (NPTS + 127) / 128), 256, 0, stream>>>(
        tbuf, W1T + (size_t)l * 256 * 128, b1 + (size_t)l * 256, ffh,
        NPTS, 256, 128);
    gemm_kernel<0><<<dim3(2, (NPTS + 127) / 128), 256, 0, stream>>>(
        ffh, W2T + (size_t)l * 128 * 256, b2 + (size_t)l * 128, ffo,
        NPTS, 128, 256);

    res_ln_kernel<false><<<(NPTS + 7) / 8, 256, 0, stream>>>(
        tbuf, ffo, nullptr, g2 + l * 128, be2 + l * 128, xbuf);
    cur = xbuf;
  }

  // BEV + two dilated convs
  fill_kernel<<<4096, 256, 0, stream>>>(canvas, (long long)BB * 128 * HH * WW);
  bev_scatter_kernel<<<NPTS, 32, 0, stream>>>(cur, coors, canvas);
  conv_kernel<1><<<BB * HH * (WW / 16), 256, 0, stream>>>(
      canvas, Wbconv, bng, bnb, bnm, bnv, bev1);
  conv_kernel<0><<<BB * HH * (WW / 16), 256, 0, stream>>>(
      bev1, Wbconv + (size_t)9 * 128 * 128, bng + 128, bnb + 128, bnm + 128,
      bnv + 128, (float*)d_out);
}
